// EvoformerEncoder_85813446574725
// MI455X (gfx1250) — matrix operand order repaired and checked by reading, not compiled, not run on hardware
//
#include <hip/hip_runtime.h>
#include <hip/hip_bf16.h>

// ---------------------------------------------------------------------------
// Evoformer encoder for MI455X (gfx1250, wave32, WMMA).
// All GEMMs but the small-N naive ones run through v_wmma_f32_16x16x32_bf16 (bf16 in, f32 accumulate).
// Fragments are loaded from LDS with contiguous b128 reads per the ISA
// 16-bit A/B VGPR layouts; staging packs fp32->bf16 pairs into dword stores.
// LayerNorm / softmax / sigmoid gating / residuals stay fp32 on VALU.
// ---------------------------------------------------------------------------

typedef __bf16 bf16_t;
typedef __attribute__((ext_vector_type(16))) __bf16 v16bf;
typedef __attribute__((ext_vector_type(8)))  __bf16 v8bf;
typedef __attribute__((ext_vector_type(8)))  float  v8f;

static constexpr int BB   = 2;
static constexpr int NN_  = 128;
static constexpr int ND   = 256;   // NODE_DIM
static constexpr int ED   = 128;   // EDGE_DIM
static constexpr int NHID = 1024;  // NODE_HID
static constexpr int EHID = 512;   // EDGE_HID
static constexpr int NH   = 8;     // heads
static constexpr int OPH  = 32;    // OP_HID
static constexpr int TMH  = 64;    // TM_HID = EDGE_DIM/2
static constexpr float LN_EPS = 1e-5f;

// ----------------------------- device helpers ------------------------------

__device__ __forceinline__ bf16_t f2bf(float x) {
  unsigned u = __builtin_bit_cast(unsigned, x);
  unsigned r = u + 0x7FFFu + ((u >> 16) & 1u);   // round-to-nearest-even
  unsigned short h = (unsigned short)(r >> 16);
  return __builtin_bit_cast(bf16_t, h);
}
// pack two floats as bf16 pair in one dword (lo = first)
__device__ __forceinline__ unsigned pack2bf(float a, float b) {
  unsigned ua = __builtin_bit_cast(unsigned, a);
  unsigned ub = __builtin_bit_cast(unsigned, b);
  ua = (ua + 0x7FFFu + ((ua >> 16) & 1u)) >> 16;
  ub = (ub + 0x7FFFu + ((ub >> 16) & 1u)) & 0xFFFF0000u;
  return (ua & 0xFFFFu) | ub;
}

// A fragment from a row-major [16 x ld] bf16 tile.
// Lane (m = lane&15, kb = 8*(lane>>4)) holds K = kb..kb+7 and 16+kb..16+kb+7:
// two contiguous 16B runs -> two ds_load_b128.
__device__ __forceinline__ v16bf frag_a_ld(const bf16_t* t, int ld) {
  int lane = threadIdx.x & 31;
  const bf16_t* p = t + (lane & 15) * ld + ((lane >> 4) << 3);
  v8bf lo = *(const v8bf*)p;
  v8bf hi = *(const v8bf*)(p + 16);
  return __builtin_shufflevector(lo, hi, 0, 1, 2, 3, 4, 5, 6, 7,
                                 8, 9, 10, 11, 12, 13, 14, 15);
}
// B fragment from an N-major [16 x ld] bf16 tile (row n holds K 0..31).
// Lane (n = lane&15, kb = 8*(lane>>4)) holds K = kb..kb+7 and 16+kb..16+kb+7.
__device__ __forceinline__ v16bf frag_b_ld(const bf16_t* t, int ld) {
  int lane = threadIdx.x & 31;
  const bf16_t* p = t + (lane & 15) * ld + ((lane >> 4) << 3);
  v8bf lo = *(const v8bf*)p;
  v8bf hi = *(const v8bf*)(p + 16);
  return __builtin_shufflevector(lo, hi, 0, 1, 2, 3, 4, 5, 6, 7,
                                 8, 9, 10, 11, 12, 13, 14, 15);
}

__device__ __forceinline__ v8f wmma_bf16(v16bf a, v16bf b, v8f c) {
  return __builtin_amdgcn_wmma_f32_16x16x32_bf16(false, a, false, b, (short)0, c,
                                                 false, false);
}

__device__ __forceinline__ float act_apply(float v, int act) {
  if (act == 1) return v > 0.f ? v : 0.f;
  if (act == 2) return 1.f / (1.f + __expf(-v));
  return v;
}

// --------------------------- generic WMMA GEMM -----------------------------
// C[b] = act(op(A[b]) @ op(B[b]) + bias) + res[b]
// 128 threads = 4 waves; block tile 64(M) x 64(N); K in 32-wide bf16 chunks.
// B tile staged once per chunk and shared by all waves; each wave reuses one
// A fragment across 4 WMMAs.
__global__ __launch_bounds__(128) void gemm_wmma_kernel(
    const float* __restrict__ A, const float* __restrict__ Bm,
    const float* __restrict__ bias, const float* __restrict__ res,
    float* __restrict__ C, int M, int N, int K,
    int lda, int ldb, int ldc,
    long long sA, long long sB, long long sC, long long sRes,
    int transA, int transB, int act) {
  __shared__ __align__(16) bf16_t Bt[64 * 32];      // [n][k] N-major
  __shared__ __align__(16) bf16_t At[4 * 16 * 32];  // per-wave [m][k]
  int tid = threadIdx.x;
  int lane = tid & 31, wave = tid >> 5;
  int nblks = (N + 63) >> 6;
  int mblk = blockIdx.x / nblks, nblk = blockIdx.x % nblks;
  int batch = blockIdx.y;
  const float* Ab = A + (long long)batch * sA;
  const float* Bb = Bm + (long long)batch * sB;
  float* Cb = C + (long long)batch * sC;
  const float* Rb = res ? (res + (long long)batch * sRes) : nullptr;
  int m0 = mblk * 64 + wave * 16;
  int n0 = nblk * 64;
  bf16_t* Aw = At + wave * 512;
  v8f acc[4] = {v8f{}, v8f{}, v8f{}, v8f{}};
  for (int k0 = 0; k0 < K; k0 += 32) {
    for (int e = tid; e < 1024; e += 128) {        // B: 64x32 = 1024 dwords
      int n = e >> 4, kk = (e & 15) << 1;
      int k = k0 + kk, ng = n0 + n;
      float v0 = 0.f, v1 = 0.f;
      if (ng < N) {
        if (k < K)
          v0 = transB ? Bb[(long long)ng * ldb + k] : Bb[(long long)k * ldb + ng];
        if (k + 1 < K)
          v1 = transB ? Bb[(long long)ng * ldb + k + 1]
                      : Bb[(long long)(k + 1) * ldb + ng];
      }
      ((unsigned*)Bt)[e] = pack2bf(v0, v1);
    }
    for (int e = lane; e < 256; e += 32) {         // A: 16x32 = 256 dwords/wave
      int rr = e >> 4, kk = (e & 15) << 1;
      int m = m0 + rr, k = k0 + kk;
      float v0 = 0.f, v1 = 0.f;
      if (m < M) {
        if (k < K)
          v0 = transA ? Ab[(long long)k * lda + m] : Ab[(long long)m * lda + k];
        if (k + 1 < K)
          v1 = transA ? Ab[(long long)(k + 1) * lda + m]
                      : Ab[(long long)m * lda + k + 1];
      }
      ((unsigned*)Aw)[e] = pack2bf(v0, v1);
    }
    __syncthreads();
    v16bf a = frag_a_ld(Aw, 32);
#pragma unroll
    for (int sub = 0; sub < 4; ++sub)
      acc[sub] = wmma_bf16(a, frag_b_ld(Bt + sub * 512, 32), acc[sub]);
    __syncthreads();
  }
  int mb = m0 + ((lane >> 4) << 3);
#pragma unroll
  for (int sub = 0; sub < 4; ++sub) {
    int n = n0 + sub * 16 + (lane & 15);
    if (n >= N) continue;
#pragma unroll
    for (int r = 0; r < 8; ++r) {
      int m = mb + r;
      if (m < M) {
        float v = acc[sub][r];
        if (bias) v += bias[n];
        v = act_apply(v, act);
        if (Rb) v += Rb[(long long)m * ldc + n];
        Cb[(long long)m * ldc + n] = v;
      }
    }
  }
}

// ---------------------- fused outer-product block (OPB) --------------------
// out[b,i,j,:] = res[b,i,j,:] + bias + sum_{c,d} a[b,i,c]*bm[b,j,d]*W[c*32+d,:]
// K-chunk t covers c=t, d=0..31: A tiles are scalar * bm-row outer products.
// Block tile: 64 pair-rows (same b,i; 4 waves x 16) x 64 out channels.
__global__ __launch_bounds__(128) void opb_kernel(
    const float* __restrict__ a, const float* __restrict__ bm,
    const float* __restrict__ W, const float* __restrict__ bias,
    const float* __restrict__ res, float* __restrict__ out,
    int N, int OH, int EDo) {
  __shared__ __align__(16) bf16_t Bt[64 * 32];      // [n][k]
  __shared__ __align__(16) bf16_t At[4 * 16 * 32];
  int tid = threadIdx.x;
  int lane = tid & 31, wave = tid >> 5;
  int nblks = EDo >> 6;
  int rblk = blockIdx.x / nblks, nblk = blockIdx.x % nblks;
  long long r0 = (long long)rblk * 64;
  int b = (int)(r0 / ((long long)N * N));
  int rem = (int)(r0 % ((long long)N * N));
  int i = rem / N;
  int j0 = rem % N;                 // 64 consecutive rows share (b,i)
  int n0 = nblk * 64;
  const float* arow = a + ((long long)b * N + i) * OH;
  bf16_t* Aw = At + wave * 512;
  v8f acc[4] = {v8f{}, v8f{}, v8f{}, v8f{}};
  for (int t = 0; t < OH; ++t) {
    float s = arow[t];
    for (int e = tid; e < 1024; e += 128) {        // W chunk [64n][32k]
      int n = e >> 4, kk = (e & 15) << 1;
      const float* wp = W + ((long long)t * 32 + kk) * EDo + n0 + n;
      ((unsigned*)Bt)[e] = pack2bf(wp[0], wp[EDo]);
    }
    for (int e = lane; e < 256; e += 32) {         // A = s * bm rows
      int rr = e >> 4, kk = (e & 15) << 1;
      int j = j0 + wave * 16 + rr;
      const float* bp = bm + ((long long)b * N + j) * OH + kk;
      ((unsigned*)Aw)[e] = pack2bf(s * bp[0], s * bp[1]);
    }
    __syncthreads();
    v16bf af = frag_a_ld(Aw, 32);
#pragma unroll
    for (int sub = 0; sub < 4; ++sub)
      acc[sub] = wmma_bf16(af, frag_b_ld(Bt + sub * 512, 32), acc[sub]);
    __syncthreads();
  }
  int mb = ((lane >> 4) << 3) + wave * 16;
#pragma unroll
  for (int sub = 0; sub < 4; ++sub) {
    int n = n0 + sub * 16 + (lane & 15);
#pragma unroll
    for (int r = 0; r < 8; ++r) {
      long long row = r0 + mb + r;
      out[row * EDo + n] = acc[sub][r] + bias[n] + res[row * EDo + n];
    }
  }
}

// --------------------------- fused attention -------------------------------
// One block per (problem p, head h). Q/K/V layout [P, L, H, dh].
// S = scale*Q.K^T + bias[p/biasPdiv, h, q, k]; softmax fp32 in LDS -> bf16 Pt;
// O = Pt.V. dh <= 32 (zero-padded). L = 128. 256 threads = 8 waves.
__global__ __launch_bounds__(256) void attn_kernel(
    const float* __restrict__ Q, const float* __restrict__ K,
    const float* __restrict__ V, const float* __restrict__ bias,
    float* __restrict__ O, int L, int H, int dh, float scale, int biasPdiv) {
  extern __shared__ char smem_raw[];
  float* S    = (float*)smem_raw;                              // L*L f32
  bf16_t* Qt  = (bf16_t*)(smem_raw + (size_t)L * L * 4);       // [l][32]
  bf16_t* Kt  = Qt + (size_t)L * 32;                           // [l][32]
  bf16_t* VtT = Kt + (size_t)L * 32;                           // [d][L] transposed
  bf16_t* Pt  = VtT + (size_t)32 * L;                          // [l][L]
  int tid = threadIdx.x;
  int p = blockIdx.x / H;
  int h = blockIdx.x % H;
  long long base = (((long long)p * L) * H + h) * dh;
  long long rstr = (long long)H * dh;
  // stage Q,K row-major (packed dword writes), V transposed
  for (int e = tid; e < L * 16; e += 256) {   // dwords of [L][32]
    int l = e >> 4, d = (e & 15) << 1;
    float q0 = 0.f, q1 = 0.f, k0v = 0.f, k1v = 0.f;
    long long off = base + (long long)l * rstr + d;
    if (d < dh)     { q0 = Q[off];     k0v = K[off]; }
    if (d + 1 < dh) { q1 = Q[off + 1]; k1v = K[off + 1]; }
    ((unsigned*)Qt)[e] = pack2bf(q0, q1);
    ((unsigned*)Kt)[e] = pack2bf(k0v, k1v);
  }
  for (int e = tid; e < L * 32; e += 256) {   // VtT[d][l]
    int l = e >> 5, d = e & 31;
    float vv = (d < dh) ? V[base + (long long)l * rstr + d] : 0.f;
    VtT[d * L + l] = f2bf(vv);
  }
  __syncthreads();

  int wave = tid >> 5, lane = tid & 31;
  int i0 = wave * 16;
  const float* bb = bias + (((long long)(p / biasPdiv) * H + h) * L) * L;

  // S = scale * Q.K^T + bias  (Kt row-major == N-major B tile for QK^T)
  v16bf qf = frag_a_ld(Qt + i0 * 32, 32);
  for (int jt = 0; jt < (L >> 4); ++jt) {
    v8f acc = {};
    acc = wmma_bf16(qf, frag_b_ld(Kt + jt * 16 * 32, 32), acc);
    int n = lane & 15, mb = (lane >> 4) << 3;
#pragma unroll
    for (int r = 0; r < 8; ++r) {
      int qi = i0 + mb + r, kj = jt * 16 + n;
      S[qi * L + kj] = acc[r] * scale + bb[qi * L + kj];
    }
  }
  __syncthreads();

  // row softmax fp32, emit packed bf16 probabilities into Pt
  if (tid < L) {
    float* row = S + tid * L;
    float mx = -1e30f;
    for (int j = 0; j < L; ++j) mx = fmaxf(mx, row[j]);
    float sum = 0.f;
    for (int j = 0; j < L; ++j) { float e = __expf(row[j] - mx); row[j] = e; sum += e; }
    float inv = 1.f / sum;
    unsigned* pr = (unsigned*)(Pt + tid * L);
    for (int j = 0; j < L; j += 2) pr[j >> 1] = pack2bf(row[j] * inv, row[j + 1] * inv);
  }
  __syncthreads();

  // O = Pt @ V  (A fragments from Pt rows; B fragments contiguous in VtT)
  for (int nt = 0; nt < 2; ++nt) {
    v8f acc = {};
    for (int k0 = 0; k0 < L; k0 += 32)
      acc = wmma_bf16(frag_a_ld(Pt + i0 * L + k0, L),
                      frag_b_ld(VtT + nt * 16 * L + k0, L), acc);
    int n = nt * 16 + (lane & 15);
    int mb = (lane >> 4) << 3;
#pragma unroll
    for (int r = 0; r < 8; ++r) {
      int l = i0 + mb + r;
      if (n < dh) O[base + (long long)l * rstr + n] = acc[r];
    }
  }
}

// ------------------------------ LayerNorm ----------------------------------
__global__ void ln_kernel(const float* __restrict__ x,
                          const float* __restrict__ g,
                          const float* __restrict__ b,
                          float* __restrict__ y, int d) {
  __shared__ float red[128];
  int row = blockIdx.x, tid = threadIdx.x;
  const float* xr = x + (long long)row * d;
  float s = 0.f, sq = 0.f;
  for (int i = tid; i < d; i += 128) { float v = xr[i]; s += v; sq += v * v; }
  red[tid] = s; __syncthreads();
  for (int o = 64; o > 0; o >>= 1) { if (tid < o) red[tid] += red[tid + o]; __syncthreads(); }
  float mean = red[0] / d;
  __syncthreads();
  red[tid] = sq; __syncthreads();
  for (int o = 64; o > 0; o >>= 1) { if (tid < o) red[tid] += red[tid + o]; __syncthreads(); }
  float var = red[0] / d - mean * mean;
  float inv = rsqrtf(var + LN_EPS);
  float* yr = y + (long long)row * d;
  for (int i = tid; i < d; i += 128) yr[i] = (xr[i] - mean) * inv * g[i] + b[i];
}

// --------------------------- small-N naive GEMM ----------------------------
__global__ void gemm_naive_kernel(const float* __restrict__ A,
                                  const float* __restrict__ W,
                                  const float* __restrict__ bias,
                                  float* __restrict__ C,
                                  long long M, int K, int N) {
  long long idx = (long long)blockIdx.x * 256 + threadIdx.x;
  if (idx >= M * N) return;
  long long m = idx / N;
  int n = (int)(idx % N);
  float s = bias ? bias[n] : 0.f;
  const float* ar = A + m * K;
  for (int k = 0; k < K; ++k) s += ar[k] * W[(long long)k * N + n];
  C[idx] = s;
}

// ------------------------------ elementwise --------------------------------
__global__ void ew_mul_kernel(const float* a, const float* b, float* c, long long n) {
  long long i = (long long)blockIdx.x * 256 + threadIdx.x;
  if (i < n) c[i] = a[i] * b[i];
}
__global__ void ew_gmuladd_kernel(const float* g, const float* x, const float* res,
                                  float* out, long long n) {
  long long i = (long long)blockIdx.x * 256 + threadIdx.x;
  if (i < n) out[i] = res[i] + g[i] * x[i];
}
__global__ void ew_add3_kernel(const float* a, const float* b, const float* c,
                               float* out, long long n) {
  long long i = (long long)blockIdx.x * 256 + threadIdx.x;
  if (i < n) out[i] = a[i] + b[i] + c[i];
}

// ------------------------------- permutes ----------------------------------
__global__ void transpose_ij_kernel(const float* in, float* out, int Bn, int N, int C) {
  long long i = (long long)blockIdx.x * 256 + threadIdx.x;
  long long tot = (long long)Bn * N * N * C;
  if (i >= tot) return;
  int c = (int)(i % C); long long t = i / C;
  int j = (int)(t % N); t /= N;
  int ii = (int)(t % N); int b = (int)(t / N);
  out[((((long long)b * N + j) * N + ii) * C) + c] = in[i];
}
__global__ void bnnc_to_bcnn_kernel(const float* in, float* out, int Bn, int N, int C) {
  long long i = (long long)blockIdx.x * 256 + threadIdx.x;
  long long tot = (long long)Bn * N * N * C;
  if (i >= tot) return;
  int c = (int)(i % C); long long t = i / C;
  int j = (int)(t % N); t /= N;
  int ii = (int)(t % N); int b = (int)(t / N);
  out[((((long long)b * C + c) * N + ii) * N) + j] = in[i];
}
__global__ void bcnn_to_bnnc_kernel(const float* in, float* out, int Bn, int N, int C) {
  long long i = (long long)blockIdx.x * 256 + threadIdx.x;
  long long tot = (long long)Bn * N * N * C;
  if (i >= tot) return;
  int c = (int)(i % C); long long t = i / C;
  int j = (int)(t % N); t /= N;
  int ii = (int)(t % N); int b = (int)(t / N);
  out[i] = in[((((long long)b * C + c) * N + ii) * N) + j];
}
__global__ void bijh_to_bhij_kernel(const float* in, float* out, int Bn, int N, int H) {
  long long i = (long long)blockIdx.x * 256 + threadIdx.x;
  long long tot = (long long)Bn * N * N * H;
  if (i >= tot) return;
  int h = (int)(i % H); long long t = i / H;
  int j = (int)(t % N); t /= N;
  int ii = (int)(t % N); int b = (int)(t / N);
  out[((((long long)b * H + h) * N + ii) * N) + j] = in[i];
}

// ------------------------------ host driver --------------------------------

struct LinP { const float *W, *b; };
struct LnP  { const float *g, *b; };
struct LayerP {
  LinP pair_bias;
  LnP sab_ln; LinP sab_q, sab_k, sab_v, sab_o;
  LnP fx_ln; LinP fx_w1, fx_w2;
  LnP opb_ln; LinP opb_a, opb_b, opb_o;
  LnP tmb_ln; LinP tmb_a, tmb_ag, tmb_bp, tmb_bg, tmb_g; LnP tmb_ln2; LinP tmb_o;
  LnP tab_ln; LinP tab_q, tab_k, tab_v, tab_bias, tab_g, tab_o;
  LnP fe_ln; LinP fe_w1, fe_w2;
};

static void gemmw(hipStream_t st, const float* A, const float* B_, const float* bias,
                  const float* res, float* C, int M, int N, int K, int act = 0,
                  int transA = 0, int transB = 0, int batch = 1,
                  long long sA = 0, long long sB = 0, long long sC = 0,
                  long long sRes = 0, int lda = 0, int ldb = 0, int ldc = 0) {
  if (!lda) lda = transA ? M : K;
  if (!ldb) ldb = transB ? K : N;
  if (!ldc) ldc = N;
  dim3 grid((unsigned)(((M + 63) / 64) * ((N + 63) / 64)), (unsigned)batch);
  gemm_wmma_kernel<<<grid, 128, 0, st>>>(A, B_, bias, res, C, M, N, K, lda, ldb, ldc,
                                         sA, sB, sC, sRes, transA, transB, act);
}
static void lnorm(hipStream_t st, const float* x, LnP p, float* y, long long rows, int d) {
  ln_kernel<<<(unsigned)rows, 128, 0, st>>>(x, p.g, p.b, y, d);
}
static void naive(hipStream_t st, const float* A, LinP p, float* C, long long M, int K, int N) {
  long long tot = M * N;
  gemm_naive_kernel<<<(unsigned)((tot + 255) / 256), 256, 0, st>>>(A, p.W, p.b, C, M, K, N);
}
static unsigned blks(long long n) { return (unsigned)((n + 255) / 256); }

extern "C" void kernel_launch(void* const* d_in, const int* in_sizes, int n_in,
                              void* d_out, int out_size, void* d_ws, size_t ws_size,
                              hipStream_t stream) {
  (void)in_sizes; (void)n_in; (void)out_size; (void)ws_size;
  // ---- parse inputs: setup_inputs() dict insertion order -------------------
  int idx = 0;
  auto nf = [&]() { return (const float*)d_in[idx++]; };
  auto lin = [&]() { LinP p; p.W = nf(); p.b = nf(); return p; };
  auto lnp = [&]() { LnP p; p.g = nf(); p.b = nf(); return p; };
  const float* Xin = nf();
  const float* Ein = nf();
  LayerP L[2];
  for (int l = 0; l < 2; ++l) {
    LayerP& p = L[l];
    p.pair_bias = lin();
    p.sab_ln = lnp(); p.sab_q = lin(); p.sab_k = lin(); p.sab_v = lin(); p.sab_o = lin();
    p.fx_ln = lnp(); p.fx_w1 = lin(); p.fx_w2 = lin();
    p.opb_ln = lnp(); p.opb_a = lin(); p.opb_b = lin(); p.opb_o = lin();
    p.tmb_ln = lnp(); p.tmb_a = lin(); p.tmb_ag = lin(); p.tmb_bp = lin();
    p.tmb_bg = lin(); p.tmb_g = lin(); p.tmb_ln2 = lnp(); p.tmb_o = lin();
    p.tab_ln = lnp(); p.tab_q = lin(); p.tab_k = lin(); p.tab_v = lin();
    p.tab_bias = lin(); p.tab_g = lin(); p.tab_o = lin();
    p.fe_ln = lnp(); p.fe_w1 = lin(); p.fe_w2 = lin();
  }
  LnP ln_X = lnp();
  LnP ln_E = lnp();

  // ---- workspace layout ----------------------------------------------------
  const size_t XN  = (size_t)BB * NN_ * ND;               // 65536
  const size_t EN  = (size_t)BB * NN_ * NN_ * ED;         // 4194304
  const size_t E64 = (size_t)BB * NN_ * NN_ * TMH;        // 2097152
  const long long BN  = (long long)BB * NN_;
  const long long BNN = (long long)BB * NN_ * NN_;
  float* w = (float*)d_ws;
  size_t off = 0;
  auto take = [&](size_t n) { float* p = w + off; off += n; return p; };
  float* XT = take(XN);
  float* Xa = take(XN); float* Xb = take(XN);
  float* XHb = take((size_t)BB * NN_ * NHID);
  float* NQ = take(XN); float* NK = take(XN); float* NV = take(XN); float* NO = take(XN);
  float* SBf  = take((size_t)BB * NN_ * NN_ * NH);
  float* BIAS = take((size_t)BB * NN_ * NN_ * NH);
  float* OA = take((size_t)BB * NN_ * OPH); float* OBf = take((size_t)BB * NN_ * OPH);
  float* Ea = take(EN); float* Eb = take(EN);
  float* Wz = take(EN); float* Wq = take(EN); float* Wk = take(EN);
  float* Wv = take(EN); float* Wo = take(EN); float* W6 = take(EN);
  float* HIDE = Wz;  // 4*EN contiguous (Wz..Wv) == BNN*EHID
  float* A0 = take(E64); float* A1 = take(E64); float* A2 = take(E64);
  float* A3 = take(E64); float* A4 = take(E64); float* A5 = take(E64);
  float* A6 = take(E64);

  const float* Xc = Xin;
  const float* Ec = Ein;
  // S (L*L f32) + Qt + Kt + VtT (3 * L*32 bf16) + Pt (L*L bf16)
  const int smem_attn = NN_ * NN_ * 4 + 3 * NN_ * 32 * 2 + NN_ * NN_ * 2;  // 122880
  const float sc_sab = 0.17677669529663687f;  // 1/sqrt(32)
  const float sc_tab = 0.25f;                 // 1/sqrt(16)
  const long long NNs = (long long)NN_ * NN_;

  for (int l = 0; l < 2; ++l) {
    LayerP& p = L[l];
    // ---- pair bias from current E: [B,N,N,H] -> [B,H,N,N] ----
    naive(stream, Ec, p.pair_bias, SBf, BNN, ED, NH);
    bijh_to_bhij_kernel<<<blks(BNN * NH), 256, 0, stream>>>(SBf, BIAS, BB, NN_, NH);
    // ---- SAB ----
    lnorm(stream, Xc, p.sab_ln, XT, BN, ND);
    gemmw(stream, XT, p.sab_q.W, p.sab_q.b, nullptr, NQ, (int)BN, ND, ND);
    gemmw(stream, XT, p.sab_k.W, p.sab_k.b, nullptr, NK, (int)BN, ND, ND);
    gemmw(stream, XT, p.sab_v.W, p.sab_v.b, nullptr, NV, (int)BN, ND, ND);
    attn_kernel<<<BB * NH, 256, smem_attn, stream>>>(NQ, NK, NV, BIAS, NO,
                                                     NN_, NH, ND / NH, sc_sab, 1);
    { float* Xn = (Xc == Xa) ? Xb : Xa;
      gemmw(stream, NO, p.sab_o.W, p.sab_o.b, Xc, Xn, (int)BN, ND, ND); Xc = Xn; }
    // ---- ffn_x ----
    lnorm(stream, Xc, p.fx_ln, XT, BN, ND);
    gemmw(stream, XT, p.fx_w1.W, p.fx_w1.b, nullptr, XHb, (int)BN, NHID, ND, 1);
    { float* Xn = (Xc == Xa) ? Xb : Xa;
      gemmw(stream, XHb, p.fx_w2.W, p.fx_w2.b, Xc, Xn, (int)BN, ND, NHID); Xc = Xn; }
    // ---- OPB ----
    lnorm(stream, Xc, p.opb_ln, XT, BN, ND);
    gemmw(stream, XT, p.opb_a.W, p.opb_a.b, nullptr, OA, (int)BN, OPH, ND);
    gemmw(stream, XT, p.opb_b.W, p.opb_b.b, nullptr, OBf, (int)BN, OPH, ND);
    { float* En = (Ec == Ea) ? Eb : Ea;
      unsigned g = (unsigned)((BNN / 64) * (ED / 64));
      opb_kernel<<<g, 128, 0, stream>>>(OA, OBf, p.opb_o.W, p.opb_o.b, Ec, En,
                                        NN_, OPH, ED);
      Ec = En; }
    // ---- TMB ----
    lnorm(stream, Ec, p.tmb_ln, Wz, BNN, ED);
    gemmw(stream, Wz, p.tmb_a.W, p.tmb_a.b, nullptr, A3, (int)BNN, TMH, ED);
    gemmw(stream, Wz, p.tmb_ag.W, p.tmb_ag.b, nullptr, A2, (int)BNN, TMH, ED, 2);
    ew_mul_kernel<<<blks(BNN * TMH), 256, 0, stream>>>(A2, A3, A0, BNN * TMH);
    gemmw(stream, Wz, p.tmb_bp.W, p.tmb_bp.b, nullptr, A3, (int)BNN, TMH, ED);
    gemmw(stream, Wz, p.tmb_bg.W, p.tmb_bg.b, nullptr, A2, (int)BNN, TMH, ED, 2);
    ew_mul_kernel<<<blks(BNN * TMH), 256, 0, stream>>>(A2, A3, A1, BNN * TMH);
    bnnc_to_bcnn_kernel<<<blks(BNN * TMH), 256, 0, stream>>>(A0, A4, BB, NN_, TMH);
    bnnc_to_bcnn_kernel<<<blks(BNN * TMH), 256, 0, stream>>>(A1, A5, BB, NN_, TMH);
    // term1 = a_plane @ b_plane^T ; term2(+res) = a_plane^T @ b_plane
    gemmw(stream, A4, A5, nullptr, nullptr, A6, NN_, NN_, NN_, 0, 0, 1,
          BB * TMH, NNs, NNs, NNs, 0, NN_, NN_, NN_);
    gemmw(stream, A4, A5, nullptr, A6, A6, NN_, NN_, NN_, 0, 1, 0,
          BB * TMH, NNs, NNs, NNs, NNs, NN_, NN_, NN_);
    bcnn_to_bnnc_kernel<<<blks(BNN * TMH), 256, 0, stream>>>(A6, A2, BB, NN_, TMH);
    lnorm(stream, A2, p.tmb_ln2, A3, BNN, TMH);
    gemmw(stream, Wz, p.tmb_g.W, p.tmb_g.b, nullptr, Wq, (int)BNN, ED, ED, 2);
    gemmw(stream, A3, p.tmb_o.W, p.tmb_o.b, nullptr, Wk, (int)BNN, ED, TMH);
    { float* En = (Ec == Ea) ? Eb : Ea;
      ew_gmuladd_kernel<<<blks((long long)EN), 256, 0, stream>>>(Wq, Wk, Ec, En, (long long)EN);
      Ec = En; }
    // ---- TAB: row pass ----
    lnorm(stream, Ec, p.tab_ln, Wz, BNN, ED);
    gemmw(stream, Wz, p.tab_q.W, p.tab_q.b, nullptr, Wq, (int)BNN, ED, ED);
    gemmw(stream, Wz, p.tab_k.W, p.tab_k.b, nullptr, Wk, (int)BNN, ED, ED);
    gemmw(stream, Wz, p.tab_v.W, p.tab_v.b, nullptr, Wv, (int)BNN, ED, ED);
    naive(stream, Wz, p.tab_bias, SBf, BNN, ED, NH);
    bijh_to_bhij_kernel<<<blks(BNN * NH), 256, 0, stream>>>(SBf, BIAS, BB, NN_, NH);
    attn_kernel<<<BB * NN_ * NH, 256, smem_attn, stream>>>(Wq, Wk, Wv, BIAS, Wo,
                                                           NN_, NH, ED / NH, sc_tab, NN_);
    gemmw(stream, Wz, p.tab_g.W, p.tab_g.b, nullptr, Wk, (int)BNN, ED, ED, 2);  // G
    gemmw(stream, Wo, p.tab_o.W, p.tab_o.b, nullptr, Wv, (int)BNN, ED, ED);     // O2
    ew_mul_kernel<<<blks((long long)EN), 256, 0, stream>>>(Wk, Wv, Wq, (long long)EN); // rowOut
    // ---- TAB: col pass (on z^T, transposed back) ----
    transpose_ij_kernel<<<blks((long long)EN), 256, 0, stream>>>(Wz, Wk, BB, NN_, ED); // zT
    gemmw(stream, Wk, p.tab_q.W, p.tab_q.b, nullptr, Wv, (int)BNN, ED, ED);  // qc
    gemmw(stream, Wk, p.tab_k.W, p.tab_k.b, nullptr, Wo, (int)BNN, ED, ED);  // kc
    gemmw(stream, Wk, p.tab_v.W, p.tab_v.b, nullptr, Wz, (int)BNN, ED, ED);  // vc
    naive(stream, Wk, p.tab_bias, SBf, BNN, ED, NH);
    bijh_to_bhij_kernel<<<blks(BNN * NH), 256, 0, stream>>>(SBf, BIAS, BB, NN_, NH);
    attn_kernel<<<BB * NN_ * NH, 256, smem_attn, stream>>>(Wv, Wo, Wz, BIAS, W6,
                                                           NN_, NH, ED / NH, sc_tab, NN_);
    gemmw(stream, Wk, p.tab_g.W, p.tab_g.b, nullptr, Wv, (int)BNN, ED, ED, 2);  // Gc
    gemmw(stream, W6, p.tab_o.W, p.tab_o.b, nullptr, Wo, (int)BNN, ED, ED);     // O2c
    ew_mul_kernel<<<blks((long long)EN), 256, 0, stream>>>(Wv, Wo, Wz, (long long)EN); // colT
    transpose_ij_kernel<<<blks((long long)EN), 256, 0, stream>>>(Wz, Wv, BB, NN_, ED); // col
    { float* En = (Ec == Ea) ? Eb : Ea;
      ew_add3_kernel<<<blks((long long)EN), 256, 0, stream>>>(Ec, Wq, Wv, En, (long long)EN);
      Ec = En; }
    // ---- ffn_e ----
    lnorm(stream, Ec, p.fe_ln, W6, BNN, ED);
    gemmw(stream, W6, p.fe_w1.W, p.fe_w1.b, nullptr, HIDE, (int)BNN, EHID, ED, 1);
    { float* En = (Ec == Ea) ? Eb : Ea;
      gemmw(stream, HIDE, p.fe_w2.W, p.fe_w2.b, Ec, En, (int)BNN, ED, EHID); Ec = En; }
  }

  // ---- final layernorms straight into d_out (X then E, flat) ----
  float* out = (float*)d_out;
  lnorm(stream, Xc, ln_X, out, BN, ND);
  lnorm(stream, Ec, ln_E, out + XN, BNN, ED);
}
